// HyenaFilter_18451179504029
// MI455X (gfx1250) — matrix operand order repaired and checked by reading, not compiled, not run on hardware
//
#include <hip/hip_runtime.h>
#include <hip/hip_bf16.h>
#include <math.h>

#define LSEQ   8192
#define DMODEL 768
#define NFFT   16384
#define NR     128
#define FO     64
#define BATCH  8

typedef __bf16 v16bf __attribute__((ext_vector_type(16)));
typedef float  v8f   __attribute__((ext_vector_type(8)));

static __device__ inline unsigned short f2bf(float f) {
    union { float f; unsigned u; } v; v.f = f;
    unsigned r = v.u + 0x7FFFu + ((v.u >> 16) & 1u);
    return (unsigned short)(r >> 16);
}
static __device__ inline __bf16 us2bf(unsigned short u) {
    return __builtin_bit_cast(__bf16, u);
}

// staged sequence layout used for async B128 copies:
//   staged[col*64 + K] = seq[col + 128*K],  col in [0,128), K in [0,64)
static __device__ inline int staged_idx(int n) {
    return (n & 127) * 64 + (n >> 7);
}

static __device__ inline v8f wmma_bf16(v16bf a, v16bf b, v8f c) {
    // D = A*B + C, f32 accumulate (v_wmma_f32_16x16x32_bf16)
    return __builtin_amdgcn_wmma_f32_16x16x32_bf16(false, a, false, b, (short)0, c, false, false);
}

// A operand (16x32 bf16): row m = lane&15 (lanes m and m+16 share row),
// half = lane>>4 selects K-subgroup.  M is row-major with leading dim ld.
static __device__ inline v16bf frag_a(const unsigned short* M, int ld, int row0,
                                      int k0, int lane, bool neg) {
    v16bf a;
    const int m = row0 + (lane & 15);
    const int half = lane >> 4;
#pragma unroll
    for (int e = 0; e < 16; ++e) {
        const int j = e >> 1, p = e & 1;
        const int kl = (j < 4) ? (half * 8 + 2 * j + p)
                               : (16 + half * 8 + 2 * (j - 4) + p);
        unsigned short u = M[m * ld + k0 + kl];
        if (neg) u ^= 0x8000u;
        a[e] = us2bf(u);
    }
    return a;
}

// B operand (32x16 bf16): col n = lane&15, K-half = lane>>4.
// M is column-major: M[col*ld + K]  (so consecutive K are adjacent).
static __device__ inline v16bf frag_b(const unsigned short* M, int ld, int col0,
                                      int k0, int lane, bool neg) {
    v16bf b;
    const int n = col0 + (lane & 15);
    const int kh = lane >> 4;
#pragma unroll
    for (int e = 0; e < 16; ++e) {
        const int j = e >> 1, p = e & 1;
        const int kl = (j < 4) ? (kh * 8 + 2 * j + p)
                               : (16 + kh * 8 + 2 * (j - 4) + p);
        unsigned short u = M[n * ld + k0 + kl];
        if (neg) u ^= 0x8000u;
        b[e] = us2bf(u);
    }
    return b;
}

// ---------------------------------------------------------------------------
// Async global->LDS stage-in (CDNA5 path, tracked by ASYNCcnt).
// Copies 8192 staged bf16 (16 KB) into the K<64 half-rows of b0r:
// chunk c (16 B = 8 elems): src byte off = c*16, lds byte off = (c>>3)*256+(c&7)*16
// ---------------------------------------------------------------------------
static __device__ inline void async_stage_in(const unsigned short* src,
                                             unsigned short* b0r_base) {
    const unsigned lds_base = (unsigned)(uintptr_t)b0r_base;  // low 32 bits = LDS offset
    const unsigned long long gbase = (unsigned long long)(uintptr_t)src;
#pragma unroll
    for (int it = 0; it < 4; ++it) {
        const int c = it * 256 + (int)threadIdx.x;
        const unsigned long long gaddr = gbase + (unsigned long long)c * 16ull;
        const unsigned laddr = lds_base + (unsigned)((c >> 3) * 256 + (c & 7) * 16);
        asm volatile("global_load_async_to_lds_b128 %0, %1, off"
                     :: "v"(laddr), "v"(gaddr) : "memory");
    }
}
static __device__ inline void wait_async0() {
    asm volatile("s_wait_asynccnt 0x0" ::: "memory");
}

// ---------------------------------------------------------------------------
// Four-step 16384-point FFT:  n = n2 + 128*n1,  k = k1 + 128*k2
//   stage1: G[k1][n2]  = sum_n1 F[k1][n1] * x[n2 + 128*n1]      (A=F, B=data)
//   twiddle: G' = G * w^(+-k1*n2)
//   stage2: X[k1+128*k2] = sum_n2 G'[k1][n2] * F[n2][k2]        (A=G', B=F)
// F[n][k] = exp(-2*pi*i*n*k/128) is symmetric -> one row-major copy serves
// both the A-gather (stage1) and the column-major B-gather (stage2).
// ---------------------------------------------------------------------------

static __device__ void init_F(unsigned short* Fre, unsigned short* Fim) {
    for (int idx = threadIdx.x; idx < NR * NR; idx += blockDim.x) {
        const int nn = idx >> 7, k = idx & 127;
        const float ph = (6.28318530717958647692f / (float)NR) * (float)((nn * k) & 127);
        float s, c;
        __sincosf(ph, &s, &c);
        Fre[idx] = f2bf(c);
        Fim[idx] = f2bf(-s);   // forward kernel e^{-i ph}
    }
}

// data buffers are complex bf16, column-major over the sequence: buf[n2*128 + n1]
static __device__ void fft_stage1(const unsigned short* bRe, const unsigned short* bIm,
                                  unsigned short* gRe, unsigned short* gIm,
                                  const unsigned short* Fre, const unsigned short* Fim,
                                  bool inv) {
    const int lane = threadIdx.x & 31;
    const int wave = threadIdx.x >> 5;
    const float tws = inv ? 1.0f : -1.0f;
    for (int tt = wave; tt < 64; tt += 8) {
        const int ti = tt >> 3, tj = tt & 7;
        v8f cre = {0.f,0.f,0.f,0.f,0.f,0.f,0.f,0.f};
        v8f cim = {0.f,0.f,0.f,0.f,0.f,0.f,0.f,0.f};
#pragma unroll
        for (int kk = 0; kk < 4; ++kk) {
            const int k0 = kk * 32;
            v16bf ar  = frag_a(Fre, NR, ti * 16, k0, lane, false);
            v16bf ai  = frag_a(Fim, NR, ti * 16, k0, lane, inv);   // conj(F) for inverse
            v16bf ani = frag_a(Fim, NR, ti * 16, k0, lane, !inv);  // -Ai
            v16bf br  = frag_b(bRe, NR, tj * 16, k0, lane, false);
            v16bf bi  = frag_b(bIm, NR, tj * 16, k0, lane, false);
            cre = wmma_bf16(ar,  br, cre);
            cre = wmma_bf16(ani, bi, cre);   // - Ai*Bi
            cim = wmma_bf16(ar,  bi, cim);
            cim = wmma_bf16(ai,  br, cim);
        }
        // twiddle and store G' row-major [k1][n2]
        const int n = tj * 16 + (lane & 15);
#pragma unroll
        for (int j = 0; j < 8; ++j) {
            const int m = ti * 16 + ((lane >> 4) * 8 + j);
            const float ph = tws * (6.28318530717958647692f / (float)NFFT) * (float)(m * n);
            float sw, cw;
            __sincosf(ph, &sw, &cw);
            const float gr = cre[j] * cw - cim[j] * sw;
            const float gi = cre[j] * sw + cim[j] * cw;
            gRe[m * NR + n] = f2bf(gr);
            gIm[m * NR + n] = f2bf(gi);
        }
    }
}

template <typename EP>
static __device__ void fft_stage2(const unsigned short* gRe, const unsigned short* gIm,
                                  const unsigned short* Fre, const unsigned short* Fim,
                                  bool inv, int maxTj, EP&& ep) {
    const int lane = threadIdx.x & 31;
    const int wave = threadIdx.x >> 5;
    const int ntiles = 8 * maxTj;
    for (int tt = wave; tt < ntiles; tt += 8) {
        const int ti = tt / maxTj, tj = tt % maxTj;
        v8f cre = {0.f,0.f,0.f,0.f,0.f,0.f,0.f,0.f};
        v8f cim = {0.f,0.f,0.f,0.f,0.f,0.f,0.f,0.f};
#pragma unroll
        for (int kk = 0; kk < 4; ++kk) {
            const int k0 = kk * 32;
            v16bf ar  = frag_a(gRe, NR, ti * 16, k0, lane, false);
            v16bf ai  = frag_a(gIm, NR, ti * 16, k0, lane, false);
            v16bf ani = frag_a(gIm, NR, ti * 16, k0, lane, true);
            v16bf br  = frag_b(Fre, NR, tj * 16, k0, lane, false); // F symmetric
            v16bf bi  = frag_b(Fim, NR, tj * 16, k0, lane, inv);   // conj for inverse
            cre = wmma_bf16(ar,  br, cre);
            cre = wmma_bf16(ani, bi, cre);
            cim = wmma_bf16(ar,  bi, cim);
            cim = wmma_bf16(ai,  br, cim);
        }
        const int n = tj * 16 + (lane & 15);
#pragma unroll
        for (int j = 0; j < 8; ++j) {
            const int m = ti * 16 + ((lane >> 4) * 8 + j);
            ep(m, n, cre[j], cim[j]);   // frequency/time index = m + 128*n
        }
    }
}

// ---------------------------------------------------------------------------
// Kernel 1: implicit filter MLP -> h (L x 64) bf16
// ---------------------------------------------------------------------------
__global__ void mlp_kernel(const float* w_in, const float* b_in, const float* freq_in,
                           const float* w_h0, const float* b_h0, const float* freq_h0,
                           const float* w_h1, const float* b_h1, const float* freq_h1,
                           unsigned short* hout) {
    const int l = blockIdx.x * blockDim.x + threadIdx.x;
    if (l >= LSEQ) return;
    const float tpos = (float)l / (float)(LSEQ - 1);
    const float wang = 6.28318530717958647692f * (float)l / (float)LSEQ;
    const float phase = 1e-4f * wang;              // bands==1 -> f = 1e-4
    const float z0 = tpos, z1 = __cosf(phase), z2 = -__sinf(phase);
    float h0[FO], h1[FO];
    for (int j = 0; j < FO; ++j) {
        float acc = b_in[j] + z0 * w_in[0 * FO + j] + z1 * w_in[1 * FO + j] + z2 * w_in[2 * FO + j];
        h0[j] = __sinf(freq_in[j] * acc);
    }
    for (int j = 0; j < FO; ++j) {
        float acc = b_h0[j];
        for (int i = 0; i < FO; ++i) acc += h0[i] * w_h0[i * FO + j];
        h1[j] = __sinf(freq_h0[j] * acc);
    }
    for (int j = 0; j < FO; ++j) {
        float acc = b_h1[j];
        for (int i = 0; i < FO; ++i) acc += h1[i] * w_h1[i * FO + j];
        hout[l * FO + j] = f2bf(__sinf(freq_h1[j] * acc));
    }
}

// ---------------------------------------------------------------------------
// Kernel 2: k = (h @ w_out) * decay, WMMA bf16, output channel-major bf16
// in the FFT-staged permutation.  grid = (L/16, 768/128), block = 256.
// ---------------------------------------------------------------------------
__global__ void filter_out_kernel(const unsigned short* h, const float* w_out,
                                  unsigned short* kT) {
    __shared__ unsigned short hA[16 * FO];     // 16 rows of h, row-major
    __shared__ unsigned short wcm[128 * FO];   // w_out slice, column-major [c][k]
    const int row0   = blockIdx.x * 16;
    const int colBlk = blockIdx.y * 128;
    for (int idx = threadIdx.x; idx < 16 * FO; idx += blockDim.x)
        hA[idx] = h[(row0 + (idx >> 6)) * FO + (idx & 63)];
    for (int idx = threadIdx.x; idx < 128 * FO; idx += blockDim.x) {
        const int c = idx >> 6, k = idx & 63;
        wcm[idx] = f2bf(w_out[k * DMODEL + colBlk + c]);
    }
    __syncthreads();
    const int lane = threadIdx.x & 31;
    const int wave = threadIdx.x >> 5;
    v8f c = {0.f,0.f,0.f,0.f,0.f,0.f,0.f,0.f};
#pragma unroll
    for (int kk = 0; kk < 2; ++kk) {
        v16bf a  = frag_a(hA,  FO, 0,         kk * 32, lane, false);
        v16bf bb = frag_b(wcm, FO, wave * 16, kk * 32, lane, false);
        c = wmma_bf16(a, bb, c);
    }
    // ExponentialModulation: deltas = linspace(ln(.01)/1.5, ln(.01)/0.3, 768)
    const float mind = 3.07011086f;   // |ln(0.01)/1.5|
    const float maxd = 15.35055433f;  // |ln(0.01)/0.3|
    const int dglob = colBlk + wave * 16 + (lane & 15);
    const float absdelta = mind + (maxd - mind) * ((float)dglob / (float)(DMODEL - 1));
    const float tscale = 1.0f / (float)(LSEQ - 1);
#pragma unroll
    for (int j = 0; j < 8; ++j) {
        const int l = row0 + ((lane >> 4) * 8 + j);
        const float dec = __expf(-(float)l * tscale * absdelta);
        kT[(size_t)dglob * LSEQ + staged_idx(l)] = f2bf(c[j] * dec);
    }
}

// ---------------------------------------------------------------------------
// Kernel 3: x (b,n,d) f32 -> xT (b,d,staged(n)) bf16
// ---------------------------------------------------------------------------
__global__ void transpose_kernel(const float* x, unsigned short* xT) {
    __shared__ float tile[32][33];
    const int n0 = blockIdx.x * 32;
    const int d0 = blockIdx.y * 32;
    const int b  = blockIdx.z;
    const int tx = threadIdx.x, ty = threadIdx.y;
#pragma unroll
    for (int r = 0; r < 4; ++r) {
        const int n = n0 + ty + r * 8;
        tile[ty + r * 8][tx] = x[((size_t)b * LSEQ + n) * DMODEL + d0 + tx];
    }
    __syncthreads();
#pragma unroll
    for (int r = 0; r < 4; ++r) {
        const int d = d0 + ty + r * 8;
        const int n = n0 + tx;
        xT[((size_t)(b * DMODEL + d)) * LSEQ + staged_idx(n)] = f2bf(tile[tx][ty + r * 8]);
    }
}

// LDS layout for FFT kernels: F(re,im) + two complex bf16 128x128 buffers = 192 KB
#define SMEM_PARTITION()                                   \
    extern __shared__ unsigned short smem[];               \
    unsigned short* Fre = smem;                            \
    unsigned short* Fim = Fre + NFFT;                      \
    unsigned short* b0r = Fim + NFFT;                      \
    unsigned short* b0i = b0r + NFFT;                      \
    unsigned short* b1r = b0i + NFFT;                      \
    unsigned short* b1i = b1r + NFFT

// zero imaginary plane and the zero-padded (K>=64) holes of the real plane;
// runs while the async DMA is in flight (disjoint LDS ranges)
static __device__ inline void zero_pad_planes(unsigned short* b0r, unsigned short* b0i) {
    for (int idx = threadIdx.x; idx < NFFT; idx += blockDim.x) {
        b0i[idx] = 0;
        if ((idx & 127) >= 64) b0r[idx] = 0;   // idx = col*128 + K, K = idx&127
    }
}

// ---------------------------------------------------------------------------
// Kernel 4: Khat[d][f] = FFT_16384(k[:,d]) / NFFT   (matches k_f = rfft(k)/N)
// ---------------------------------------------------------------------------
__global__ void khat_kernel(const unsigned short* kT, float2* Khat) {
    SMEM_PARTITION();
    const int d = blockIdx.x;
    async_stage_in(kT + (size_t)d * LSEQ, b0r);   // ASYNCcnt-tracked DMA to LDS
    zero_pad_planes(b0r, b0i);
    init_F(Fre, Fim);
    wait_async0();
    __syncthreads();
    fft_stage1(b0r, b0i, b1r, b1i, Fre, Fim, false);
    __syncthreads();
    const float sc = 1.0f / (float)NFFT;
    float2* Kd = Khat + (size_t)d * NFFT;
    fft_stage2(b1r, b1i, Fre, Fim, false, 8,
        [&](int m, int n, float re, float im) {
            Kd[m + NR * n] = make_float2(re * sc, im * sc);
        });
}

// ---------------------------------------------------------------------------
// Kernel 5: fused  y[b,:,d] = Re(IFFT( FFT(x[b,:,d]) * Khat[d] )) + bias[d]
// one workgroup per (b,d); inverse stage2 truncated to t < 8192 (maxTj = 4)
// ---------------------------------------------------------------------------
__global__ void conv_kernel(const unsigned short* xT, const float2* Khat,
                            const float* bias, float* out) {
    SMEM_PARTITION();
    const int bd = blockIdx.x;
    const int b = bd / DMODEL, d = bd % DMODEL;
    async_stage_in(xT + (size_t)(b * DMODEL + d) * LSEQ, b0r);
    zero_pad_planes(b0r, b0i);
    init_F(Fre, Fim);
    __builtin_prefetch(&Khat[(size_t)d * NFFT], 0, 0);
    wait_async0();
    __syncthreads();
    fft_stage1(b0r, b0i, b1r, b1i, Fre, Fim, false);     // forward
    __syncthreads();
    const float2* Kd = Khat + (size_t)d * NFFT;
    // forward stage2 fused with pointwise multiply; result stored so that the
    // inverse stage1 reads it as a col-major sequence: index k1*128 + k2
    fft_stage2(b1r, b1i, Fre, Fim, false, 8,
        [&](int m, int n, float re, float im) {
            const float2 kf = Kd[m + NR * n];
            const float xr = re * kf.x - im * kf.y;
            const float xi = re * kf.y + im * kf.x;
            b0r[m * NR + n] = f2bf(xr);
            b0i[m * NR + n] = f2bf(xi);
        });
    __syncthreads();
    fft_stage1(b0r, b0i, b1r, b1i, Fre, Fim, true);      // inverse
    __syncthreads();
    const float bs = bias[d];
    const float inn = 1.0f / (float)NFFT;                // irfft backward norm
    fft_stage2(b1r, b1i, Fre, Fim, true, 4,
        [&](int m, int n, float re, float im) {
            (void)im;
            const int t = m + NR * n;                    // t < 8192 since n < 64
            out[((size_t)b * LSEQ + t) * DMODEL + d] = re * inn + bs;
        });
}

extern "C" void kernel_launch(void* const* d_in, const int* in_sizes, int n_in,
                              void* d_out, int out_size, void* d_ws, size_t ws_size,
                              hipStream_t stream) {
    (void)in_sizes; (void)n_in; (void)out_size; (void)ws_size;
    const float* x       = (const float*)d_in[0];
    const float* w_in    = (const float*)d_in[1];
    const float* b_in    = (const float*)d_in[2];
    const float* freq_in = (const float*)d_in[3];
    const float* w_h0    = (const float*)d_in[4];
    const float* b_h0    = (const float*)d_in[5];
    const float* freq_h0 = (const float*)d_in[6];
    const float* w_h1    = (const float*)d_in[7];
    const float* b_h1    = (const float*)d_in[8];
    const float* freq_h1 = (const float*)d_in[9];
    const float* w_out   = (const float*)d_in[10];
    const float* bias    = (const float*)d_in[11];
    float* out = (float*)d_out;

    // workspace layout (bytes):
    //   h   bf16 : 8192*64*2              =   1,048,576
    //   xT  bf16 : 8*768*8192*2           = 100,663,296
    //   kT  bf16 : 768*8192*2             =  12,582,912
    //   Khat f32x2: 768*16384*8           = 100,663,296
    char* ws = (char*)d_ws;
    unsigned short* h_ws = (unsigned short*)ws;
    unsigned short* xT   = (unsigned short*)(ws + 1048576ull);
    unsigned short* kT   = (unsigned short*)(ws + 1048576ull + 100663296ull);
    float2*         Khat = (float2*)(ws + 1048576ull + 100663296ull + 12582912ull);

    mlp_kernel<<<LSEQ / 256, 256, 0, stream>>>(w_in, b_in, freq_in,
                                               w_h0, b_h0, freq_h0,
                                               w_h1, b_h1, freq_h1, h_ws);
    filter_out_kernel<<<dim3(LSEQ / 16, DMODEL / 128), 256, 0, stream>>>(h_ws, w_out, kT);
    transpose_kernel<<<dim3(LSEQ / 32, DMODEL / 32, BATCH), dim3(32, 8), 0, stream>>>(x, xT);

    const size_t smem = 6ull * NFFT * sizeof(unsigned short);   // 192 KB of the 320 KB WGP LDS
    khat_kernel<<<DMODEL, 256, smem, stream>>>(kT, Khat);
    conv_kernel<<<BATCH * DMODEL, 256, smem, stream>>>(xT, Khat, bias, out);
}
